// RSCL_39127152066701
// MI455X (gfx1250) — matrix operand order repaired and checked by reading, not compiled, not run on hardware
//
#include <hip/hip_runtime.h>
#include <hip/hip_bf16.h>
#include <math.h>

typedef __attribute__((ext_vector_type(16))) _Float16 v16h;
typedef __attribute__((ext_vector_type(8)))  float    v8f;
typedef __attribute__((ext_vector_type(4)))  int      v4i;

#define C_NUM   19
#define D_NUM   128
#define EPS_V   1e-7f
#define BETA_V  0.7f
#define EMA_V   0.999f
// logits scale 1/TAU_HARD = 5.0 folded into normalized features

#if __has_builtin(__builtin_amdgcn_global_load_async_to_lds_b128)
#define HAVE_ASYNC_LDS 1
#else
#define HAVE_ASYNC_LDS 0
#endif

#if HAVE_ASYNC_LDS
typedef __attribute__((address_space(1))) v4i as1_v4i;   // global int4
typedef __attribute__((address_space(3))) v4i as3_v4i;   // LDS int4
__device__ __forceinline__ as1_v4i* to_glb(const void* p) {
    return (as1_v4i*)(void*)p;
}
__device__ __forceinline__ as3_v4i* to_lds(void* p) {
    return (as3_v4i*)p;
}
#endif

__device__ __forceinline__ void wait_async_zero() {
#if __has_builtin(__builtin_amdgcn_s_wait_asynccnt)
    __builtin_amdgcn_s_wait_asynccnt(0);
#else
    asm volatile("s_wait_asynccnt 0x0" ::: "memory");
#endif
}

// ---------------------------------------------------------------------------
// Kernel 1: UCPS loss + reliability map r_u + consensus map.
// One thread per hi-res pixel (B*H*W = 1,048,576). Coalesced per-channel reads.
// ---------------------------------------------------------------------------
__global__ __launch_bounds__(256) void ucps_kernel(
    const float* __restrict__ lu1, const float* __restrict__ lu2,
    float* __restrict__ rU, int* __restrict__ cons, float* __restrict__ ucpsAcc)
{
    int n = blockIdx.x * 256 + threadIdx.x;        // < 1048576
    int b   = n >> 18;                              // / (512*512)
    int pix = n & 262143;
    size_t base = (size_t)b * (C_NUM * 262144) + pix;

    float v1[C_NUM], v2[C_NUM];
#pragma unroll
    for (int c = 0; c < C_NUM; ++c) {
        v1[c] = lu1[base + (size_t)c * 262144];
        v2[c] = lu2[base + (size_t)c * 262144];
    }

    int y1 = 0, y2 = 0;
    float m1 = v1[0], m2 = v2[0];
#pragma unroll
    for (int c = 1; c < C_NUM; ++c) {
        if (v1[c] > m1) { m1 = v1[c]; y1 = c; }
        if (v2[c] > m2) { m2 = v2[c]; y2 = c; }
    }

    float s1 = 0.f, s2 = 0.f;
#pragma unroll
    for (int c = 0; c < C_NUM; ++c) {
        s1 += expf(v1[c] - m1);
        s2 += expf(v2[c] - m2);
    }
    float is1 = 1.f / s1, is2 = 1.f / s2;

    float kl = 0.f, selA = 0.f, selB = 0.f;
#pragma unroll
    for (int c = 0; c < C_NUM; ++c) {
        float q1 = expf(v1[c] - m1) * is1;
        float q2 = expf(v2[c] - m2) * is2;
        float mm = 0.5f * (q1 + q2);
        float lm = logf(fmaxf(mm, EPS_V));
        kl += q1 * (logf(fmaxf(q1, EPS_V)) - lm)
            + q2 * (logf(fmaxf(q2, EPS_V)) - lm);
        selA = (c == y2) ? v1[c] : selA;   // logit_u1 at y2
        selB = (c == y1) ? v2[c] : selB;   // logit_u2 at y1
    }

    float conf = 0.5f * (is1 + is2);        // max softmax prob = exp(0)/s
    float r = (y1 == y2) ? conf * expf(-0.5f * kl) : 0.f;
    float ce1 = (m1 + logf(s1)) - selA;
    float ce2 = (m2 + logf(s2)) - selB;
    float contrib = r * (ce1 + ce2);

    rU[n]   = r;
    cons[n] = (y1 != y2) ? 0 : y1;

    // wave32 reduction, one atomic per wave
#pragma unroll
    for (int off = 1; off < 32; off <<= 1)
        contrib += __shfl_xor(contrib, off, 32);
    if ((threadIdx.x & 31) == 0) atomicAdd(ucpsAcc, contrib);
}

// ---------------------------------------------------------------------------
// Kernel 2: nearest/bilinear downsamples + weighted class prototypes
// (LDS segment-sum per block, then global atomics).
// One thread per low-res pixel (65536).
// ---------------------------------------------------------------------------
__global__ __launch_bounds__(256) void proto_kernel(
    const float* __restrict__ fl1, const float* __restrict__ fl2,
    const float* __restrict__ fu1, const float* __restrict__ fu2,
    const int* __restrict__ gt, const float* __restrict__ rU,
    const int* __restrict__ cons,
    const int* __restrict__ curIter, const int* __restrict__ maxIter,
    int* __restrict__ gtD, int* __restrict__ psD,
    float* __restrict__ gsL, float* __restrict__ gcL,
    float* __restrict__ gsU, float* __restrict__ gwU, float* __restrict__ gcU)
{
    __shared__ float smem[C_NUM * D_NUM * 2 + 3 * C_NUM];  // 4921 floats
    float* sL = smem;
    float* sU = sL + C_NUM * D_NUM;
    float* cL = sU + C_NUM * D_NUM;
    float* wU = cL + C_NUM;
    float* cU = wU + C_NUM;

    int tid = threadIdx.x;
    for (int i = tid; i < C_NUM * D_NUM * 2 + 3 * C_NUM; i += 256) smem[i] = 0.f;
    __syncthreads();

    int n  = blockIdx.x * 256 + tid;      // < 65536
    int b  = n >> 14;
    int sp = n & 16383;
    int i  = sp >> 7, j = sp & 127;
    int y0 = 4 * i, x0 = 4 * j;
    int hi = b * 262144 + y0 * 512 + x0;

    int g  = gt[hi];
    int ps = cons[hi];
    gtD[n] = g;
    psD[n] = ps;

    // bilinear 512->128, half-pixel centers => 2x2 average at rows/cols +1,+2
    int rb = b * 262144;
    float rd = 0.25f * (rU[rb + (y0 + 1) * 512 + x0 + 1] + rU[rb + (y0 + 1) * 512 + x0 + 2]
                      + rU[rb + (y0 + 2) * 512 + x0 + 1] + rU[rb + (y0 + 2) * 512 + x0 + 2]);

    float ratio = (float)curIter[0] / (float)((maxIter[0] > 1) ? maxIter[0] : 1);
    float tauHigh = 0.85f - (0.85f - 0.5f) * cosf(3.14159265358979f * ratio * 0.5f);
    float ar = (rd > tauHigh) ? rd : 0.f;

    atomicAdd(&cL[g], 1.f);
    atomicAdd(&cU[ps], 1.f);
    atomicAdd(&wU[ps], ar);

    size_t fb = (size_t)b * (D_NUM * 16384) + sp;
    for (int d = 0; d < D_NUM; ++d) {
        size_t a = fb + (size_t)d * 16384;
        float flv = 0.5f * (fl1[a] + fl2[a]);
        float fuv = 0.5f * (fu1[a] + fu2[a]);
        atomicAdd(&sL[g * D_NUM + d], flv);
        atomicAdd(&sU[ps * D_NUM + d], ar * fuv);
    }
    __syncthreads();

    for (int k = tid; k < C_NUM * D_NUM; k += 256) {
        atomicAdd(&gsL[k], sL[k]);
        atomicAdd(&gsU[k], sU[k]);
    }
    for (int k = tid; k < C_NUM; k += 256) {
        atomicAdd(&gcL[k], cL[k]);
        atomicAdd(&gwU[k], wU[k]);
        atomicAdd(&gcU[k], cU[k]);
    }
}

// ---------------------------------------------------------------------------
// Kernel 3: merge prototypes, l2norm, EMA memory update; emit f16 memory
// (padded to 32 classes) and a float valid-mask for the CE weighting.
// ---------------------------------------------------------------------------
__global__ void mem_kernel(
    const float* __restrict__ sL, const float* __restrict__ cL,
    const float* __restrict__ sU, const float* __restrict__ wU,
    const float* __restrict__ cU,
    const float* __restrict__ memOld, const unsigned char* __restrict__ memInit,
    float* __restrict__ initF, _Float16* __restrict__ memh)
{
    int c = threadIdx.x;
    if (c >= 32) return;
    if (c >= C_NUM) {
        initF[c] = 0.f;
        for (int d = 0; d < D_NUM; ++d) memh[c * D_NUM + d] = (_Float16)0.f;
        return;
    }
    bool presL = cL[c] >= 1.f;
    bool presU = cU[c] >= 1.f;
    bool pres  = presL || presU;
    bool both  = presL && presU;
    bool oldI  = memInit[c] != 0;

    float icl = 1.f / (cL[c] + EPS_V);
    float iwu = 1.f / (wU[c] + EPS_V);

    float ss = 0.f;
    for (int d = 0; d < D_NUM; ++d) {
        float pl = sL[c * D_NUM + d] * icl;
        float pu = sU[c * D_NUM + d] * iwu;
        float m  = both ? (BETA_V * pl + (1.f - BETA_V) * pu) : (presL ? pl : pu);
        ss += m * m;
    }
    float inv = 1.f / fmaxf(sqrtf(ss), 1e-12f);

    float ss2 = 0.f;
    for (int d = 0; d < D_NUM; ++d) {
        float pl = sL[c * D_NUM + d] * icl;
        float pu = sU[c * D_NUM + d] * iwu;
        float m  = both ? (BETA_V * pl + (1.f - BETA_V) * pu) : (presL ? pl : pu);
        float pn = m * inv;
        float e  = EMA_V * memOld[c * D_NUM + d] + (1.f - EMA_V) * pn;
        ss2 += e * e;
    }
    float inv2 = 1.f / fmaxf(sqrtf(ss2), 1e-12f);

    for (int d = 0; d < D_NUM; ++d) {
        float pl = sL[c * D_NUM + d] * icl;
        float pu = sU[c * D_NUM + d] * iwu;
        float m  = both ? (BETA_V * pl + (1.f - BETA_V) * pu) : (presL ? pl : pu);
        float pn = m * inv;
        float e  = (EMA_V * memOld[c * D_NUM + d] + (1.f - EMA_V) * pn) * inv2;
        float nm = (pres && !oldI) ? pn : ((pres && oldI) ? e : memOld[c * D_NUM + d]);
        memh[c * D_NUM + d] = (_Float16)nm;
    }
    initF[c] = (oldI || pres) ? 1.f : 0.f;
}

// ---------------------------------------------------------------------------
// Kernel 4: hard-contrastive CE via WMMA.
// Per wave: 16 pixels x 19 classes, K=128 -> 8x v_wmma_f32_16x16x32_f16.
// A = f16 memory (classes, two 16-row tiles), B = l2norm(feat)/TAU in f16.
// Staging uses GLOBAL_LOAD_ASYNC_TO_LDS_B128 (ASYNCcnt) when available.
// D layout: pixel n in lanes {n, n+16}; classes split acc0(0..15)/acc1(16..18).
// ---------------------------------------------------------------------------
__global__ __launch_bounds__(128) void hard_kernel(
    const float* __restrict__ fl1, const float* __restrict__ fl2,
    const float* __restrict__ fu1, const float* __restrict__ fu2,
    const int* __restrict__ gtD, const int* __restrict__ psD,
    const _Float16* __restrict__ memh, const float* __restrict__ initF,
    float* __restrict__ sumCE, float* __restrict__ sumV)
{
    __shared__ _Float16 sA[32 * D_NUM];      // f16 memory tile (shared by block)
    __shared__ float    sF[4][D_NUM * 16];   // per-wave pixel tile [d][p]
    __shared__ float    sInv[4][16];

    int tid  = threadIdx.x;
    int wave = tid >> 5;
    int lane = tid & 31;
    int work = blockIdx.x * 4 + wave;        // 0..16383
    int feat = work >> 12;                   // 4096 tiles per feature tensor
    int tile = work & 4095;

    const float* fp = (feat == 0) ? fl1 : (feat == 1) ? fl2 : (feat == 2) ? fu1 : fu2;
    const int*   lb = (feat < 2) ? gtD : psD;

    int pixBase = tile * 16;
    int b   = pixBase >> 14;
    int sp0 = pixBase & 16383;
    const float* base = fp + (size_t)b * (D_NUM * 16384) + sp0;
    float* F = sF[wave];

#if HAVE_ASYNC_LDS
    // f16 memory tile: 8KB contiguous, block-cooperative async copy (16B/lane)
#pragma unroll
    for (int it = 0; it < 4; ++it) {
        int h0 = (tid + 128 * it) * 8;       // half index, 16 bytes per lane
        __builtin_amdgcn_global_load_async_to_lds_b128(
            to_glb(memh + h0), to_lds(&sA[h0]), 0, 0);
    }
    // per-wave feature tile: 8KB, 16 async B128 per wave
#pragma unroll
    for (int it = 0; it < 16; ++it) {
        int f0 = (lane + 32 * it) * 4;       // float index (4 floats = 16B)
        int d = f0 >> 4, p = f0 & 15;
        __builtin_amdgcn_global_load_async_to_lds_b128(
            to_glb(base + (size_t)d * 16384 + p), to_lds(&F[f0]), 0, 0);
    }
    wait_async_zero();
#else
    for (int k = tid; k < 32 * D_NUM; k += 128) sA[k] = memh[k];
    for (int k = lane; k < D_NUM * 16; k += 32) {
        int d = k >> 4, p = k & 15;
        F[k] = base[(size_t)d * 16384 + p];
    }
#endif
    __syncthreads();

    if (lane < 16) {
        float ss = 0.f;
        for (int d = 0; d < D_NUM; ++d) { float v = F[d * 16 + lane]; ss += v * v; }
        float nm = fmaxf(sqrtf(ss), 1e-12f);
        sInv[wave][lane] = (1.f / nm) * 5.0f;   // fold 1/TAU_HARD
    }
    __syncthreads();

    int p  = lane & 15;
    int hi = lane >> 4;              // 0: K 0-7,16-23 / classes m, 1: K 8-15,24-31 / m+8
    float inv = sInv[wave][p];

    v8f acc0 = {};
    v8f acc1 = {};
#pragma unroll
    for (int k0 = 0; k0 < D_NUM; k0 += 32) {
        v16h bf;
#pragma unroll
        for (int e = 0; e < 16; ++e) {
            int K = k0 + ((e < 8) ? (hi * 8 + e) : (16 + hi * 8 + (e - 8)));
            bf[e] = (_Float16)(F[K * 16 + p] * inv);
        }
        v16h a0, a1;
#pragma unroll
        for (int v = 0; v < 8; ++v) {
            int kl = (v < 4) ? (2 * v) : (16 + 2 * (v - 4));
            int K = k0 + hi * 8 + kl;
            a0[2 * v]     = sA[p * D_NUM + K];
            a0[2 * v + 1] = sA[p * D_NUM + K + 1];
            a1[2 * v]     = sA[(p + 16) * D_NUM + K];
            a1[2 * v + 1] = sA[(p + 16) * D_NUM + K + 1];
        }
        acc0 = __builtin_amdgcn_wmma_f32_16x16x32_f16(false, a0, false, bf, (short)0, acc0, false, false);
        acc1 = __builtin_amdgcn_wmma_f32_16x16x32_f16(false, a1, false, bf, (short)0, acc1, false, false);
    }

    int lab = lb[pixBase + p];
    float valid = initF[lab];

    // local max + selected logit over owned classes
    float mx = -1e30f, sel = 0.f;
#pragma unroll
    for (int v = 0; v < 8; ++v) {
        float val = acc0[v];
        int c = (hi ? 8 : 0) + v;
        mx = fmaxf(mx, val);
        sel += (c == lab) ? val : 0.f;
    }
#pragma unroll
    for (int v = 0; v < 3; ++v) {               // classes 16..18 (hi=0 lanes only)
        float val = hi ? -1e30f : acc1[v];
        int c = 16 + v;
        mx = fmaxf(mx, val);
        sel += (!hi && c == lab) ? val : 0.f;
    }
    mx = fmaxf(mx, __shfl_xor(mx, 16, 32));

    float se = 0.f;
#pragma unroll
    for (int v = 0; v < 8; ++v) se += expf(acc0[v] - mx);
#pragma unroll
    for (int v = 0; v < 3; ++v) se += hi ? 0.f : expf(acc1[v] - mx);
    se  += __shfl_xor(se, 16, 32);
    sel += __shfl_xor(sel, 16, 32);

    float ce = (mx + logf(se)) - sel;           // -log_softmax at label
    float cce = hi ? 0.f : valid * ce;          // count each pixel once
    float cv  = hi ? 0.f : valid;
#pragma unroll
    for (int off = 1; off < 32; off <<= 1) {
        cce += __shfl_xor(cce, off, 32);
        cv  += __shfl_xor(cv,  off, 32);
    }
    if (lane == 0) {
        atomicAdd(&sumCE[feat], cce);
        atomicAdd(&sumV[feat],  cv);
    }
}

// ---------------------------------------------------------------------------
// Kernel 5: finalize losses.
// ---------------------------------------------------------------------------
__global__ void final_kernel(const float* __restrict__ ucps,
                             const float* __restrict__ sumCE,
                             const float* __restrict__ sumV,
                             float* __restrict__ out)
{
    if (threadIdx.x == 0 && blockIdx.x == 0) {
        float h[4];
        for (int i = 0; i < 4; ++i)
            h[i] = (sumV[i] > 0.f) ? sumCE[i] / fmaxf(sumV[i], 1.f) : 0.f;
        float hl = 0.5f * (h[0] + h[1]);
        float hu = 0.5f * (h[2] + h[3]);
        out[0] = hl + hu;                         // loss_dgpc
        out[1] = ucps[0] / 1048576.0f;            // loss_ucps (mean over B*H*W)
    }
}

// ---------------------------------------------------------------------------
extern "C" void kernel_launch(void* const* d_in, const int* in_sizes, int n_in,
                              void* d_out, int out_size, void* d_ws, size_t ws_size,
                              hipStream_t stream) {
    const float* fl1 = (const float*)d_in[0];
    const float* fl2 = (const float*)d_in[1];
    const float* fu1 = (const float*)d_in[2];
    const float* fu2 = (const float*)d_in[3];
    // d_in[4], d_in[5] (logits_l1/l2) are unused by the reference
    const float* lu1 = (const float*)d_in[6];
    const float* lu2 = (const float*)d_in[7];
    const float* memOld = (const float*)d_in[8];
    const int*   gt  = (const int*)d_in[9];
    const unsigned char* memInit = (const unsigned char*)d_in[10];
    const int* curIter = (const int*)d_in[11];
    const int* maxIter = (const int*)d_in[12];
    float* out = (float*)d_out;

    float* ws = (float*)d_ws;
    size_t o = 0;
    float* rU  = ws;                o += 1048576;              // [B,H,W] reliability
    int*   cons = (int*)(ws + o);   o += 1048576;              // consensus map
    int*   gtD  = (int*)(ws + o);   o += 65536;                // gt downsampled
    int*   psD  = (int*)(ws + o);   o += 65536;                // pseudo downsampled
    float* zbase = ws + o;                                     // zeroed accumulators:
    float* sL   = zbase;                                       // 2432
    float* cL   = sL + C_NUM * D_NUM;                          // 19
    float* sU   = cL + C_NUM;                                  // 2432
    float* wU   = sU + C_NUM * D_NUM;                          // 19
    float* cU   = wU + C_NUM;                                  // 19
    float* ucps = cU + C_NUM;                                  // 1
    float* sumCE = ucps + 1;                                   // 4
    float* sumV  = sumCE + 4;                                  // 4
    o += 2 * C_NUM * D_NUM + 3 * C_NUM + 9;
    float* initF = ws + o;          o += 32;
    _Float16* memh = (_Float16*)(ws + o); o += 2048;           // 32x128 f16

    (void)hipMemsetAsync(zbase, 0, (2 * C_NUM * D_NUM + 3 * C_NUM + 9) * sizeof(float), stream);

    ucps_kernel <<<4096, 256, 0, stream>>>(lu1, lu2, rU, cons, ucps);
    proto_kernel<<<256,  256, 0, stream>>>(fl1, fl2, fu1, fu2, gt, rU, cons,
                                           curIter, maxIter, gtD, psD,
                                           sL, cL, sU, wU, cU);
    mem_kernel  <<<1,    32,  0, stream>>>(sL, cL, sU, wU, cU, memOld, memInit,
                                           initF, memh);
    hard_kernel <<<4096, 128, 0, stream>>>(fl1, fl2, fu1, fu2, gtD, psD,
                                           memh, initF, sumCE, sumV);
    final_kernel<<<1,    32,  0, stream>>>(ucps, sumCE, sumV, out);
}
